// MultiHeadAttention_63015760167496
// MI455X (gfx1250) — matrix operand order repaired and checked by reading, not compiled, not run on hardware
//
#include <hip/hip_runtime.h>
#include <hip/hip_bf16.h>

typedef __attribute__((ext_vector_type(16))) __bf16 v16bf;
typedef __attribute__((ext_vector_type(8)))  __bf16 v8bf;
typedef __attribute__((ext_vector_type(8)))  float  v8f;
typedef unsigned int u32x4 __attribute__((ext_vector_type(4)));
typedef int          i32x8 __attribute__((ext_vector_type(8)));
typedef int          i32x4 __attribute__((ext_vector_type(4)));

static __device__ __forceinline__ v8f wmma_bf16(v16bf a, v16bf b, v8f c) {
  // D = A(16x32) * B(32x16) + C, f32 accumulate
  return __builtin_amdgcn_wmma_f32_16x16x32_bf16(false, a, false, b, (short)0, c, false, false);
}

// Pack two contiguous 16B halves into a v16bf fragment register set.
static __device__ __forceinline__ v16bf pack16(v8bf lo, v8bf hi) {
  v16bf r;
#pragma unroll
  for (int e = 0; e < 8; ++e) { r[e] = lo[e]; r[e + 8] = hi[e]; }
  return r;
}

// B-fragment: 16 contiguous bf16 (two 16B loads; 16B alignment only required)
static __device__ __forceinline__ v16bf load_b16(const void* p) {
  return pack16(*(const v8bf*)p, *(const v8bf*)((const char*)p + 16));
}

// A-fragment (16x32 bf16 MxK): lane holds row (lane&15); kgroup = lane>>4.
// element e -> k = kbase + (e<8 ? e : e+8) + 8*kgroup  => two contiguous 16B runs.
static __device__ __forceinline__ v16bf load_a_frag(const __bf16* __restrict__ row,
                                                    int kbase, int kg) {
  return pack16(*(const v8bf*)(row + kbase + 8 * kg),
                *(const v8bf*)(row + kbase + 8 * kg + 16));
}

// Async global -> LDS copy, 16B per lane, tracked by ASYNCcnt.
static __device__ __forceinline__ void async_b128(unsigned ldsOff, const void* ga) {
  asm volatile("global_load_async_to_lds_b128 %0, %1, off"
               :: "v"(ldsOff), "v"((unsigned long long)(uintptr_t)ga)
               : "memory");
}
static __device__ __forceinline__ void wait_async0() {
  asm volatile("s_wait_asynccnt 0" ::: "memory");
}
static __device__ __forceinline__ void wait_async4() {
  asm volatile("s_wait_asynccnt 4" ::: "memory");   // async ops complete in order
}

// Tensor Data Mover: 2D tile (tileW elems x tileH rows, 2B elems) global -> LDS,
// with hardware row padding into the LDS pitch. D# packing per ISA 08_async_tensor.
// padIntervalCode: pad after 2<<code dwords; padAmountCode: (code+1) dwords inserted.
static __device__ __forceinline__ void tdm_load_2d(unsigned ldsOff, const void* gaddr,
                                                   unsigned tileW, unsigned tileH,
                                                   unsigned strideElems,
                                                   unsigned padIntervalCode,
                                                   unsigned padAmountCode) {
  const unsigned long long ga = (unsigned long long)(uintptr_t)gaddr;
  const unsigned td0 = strideElems;   // tensor_dim0 >= tileW (no OOB in-tile)
  const unsigned td1 = 1u << 20;      // tensor_dim1 >> tileH
  u32x4 g0;
  g0[0] = 1u;                                                   // count=1 (valid, user)
  g0[1] = ldsOff;                                               // lds_addr (bytes)
  g0[2] = (unsigned)(ga & 0xFFFFFFFFu);                         // global_addr[31:0]
  g0[3] = (unsigned)((ga >> 32) & 0x01FFFFFFu) | 0x80000000u;   // addr[56:32] | type=2
  i32x8 g1;
  g1[0] = (int)((1u << 16) | (1u << 20) |                       // data_size=2B, pad_en
                (padIntervalCode << 22) | (padAmountCode << 25));
  g1[1] = (int)((td0 & 0xFFFFu) << 16);                         // tensor_dim0 lo
  g1[2] = (int)((td0 >> 16) | ((td1 & 0xFFFFu) << 16));         // dim0 hi | dim1 lo
  g1[3] = (int)((td1 >> 16) | (tileW << 16));                   // dim1 hi | tile_dim0
  g1[4] = (int)tileH;                                           // tile_dim1 (tile_dim2=0)
  g1[5] = (int)strideElems;                                     // tensor_dim0_stride lo32
  g1[6] = 0;                                                    // stride hi | dim1_stride
  g1[7] = 0;
  const i32x4 z4 = {};
  const i32x8 z8 = {};
  // 6-arg toolchain signature: (g0, g1, g2, g3, g4, cpol)
  __builtin_amdgcn_tensor_load_to_lds(g0, g1, z4, z4, z8, 0);
}

// Butterfly xor-shuffle within 16-lane groups via DS_SWIZZLE (no bpermute clamping VALU).
template <int XMASK>
static __device__ __forceinline__ float swz_xor(float v) {
  return __int_as_float(
      __builtin_amdgcn_ds_swizzle(__float_as_int(v), (XMASK << 10) | 0x1f));
}
static __device__ __forceinline__ float red_max16(float v) {
  v = fmaxf(v, swz_xor<8>(v)); v = fmaxf(v, swz_xor<4>(v));
  v = fmaxf(v, swz_xor<2>(v)); v = fmaxf(v, swz_xor<1>(v));
  return v;
}
static __device__ __forceinline__ float red_sum16(float v) {
  v += swz_xor<8>(v); v += swz_xor<4>(v);
  v += swz_xor<2>(v); v += swz_xor<1>(v);
  return v;
}

// ---------------- fp32 -> bf16 converters ----------------
__global__ void cvt_bf16(const float* __restrict__ s, __bf16* __restrict__ d, int n) {
  int i = blockIdx.x * 256 + threadIdx.x;
  if (i < n) d[i] = (__bf16)s[i];
}

// dst[c*R + r] = src[r*C + c]   (writes coalesced; weights become N x K row-major)
__global__ void cvtT_bf16(const float* __restrict__ s, __bf16* __restrict__ d, int R, int C) {
  int i = blockIdx.x * 256 + threadIdx.x;
  if (i < R * C) {
    int r = i % R, c = i / R;
    d[i] = (__bf16)s[(size_t)r * C + c];
  }
}

// ---------------- bf16 WMMA GEMM:  C[MxN] = A[MxK] * Bt[NxK]^T ----------------
// 256 threads = 8 waves (2 x 4); block tile 128x128; wave tile 64x32; K-step 32.
// A/B K-slices are double-buffered in LDS via async global->LDS (4 b128/thread/step).
// LDS: A bufs 2 x 128 rows x 80B (0..20480), B bufs 2 x 128 rows x 80B (20480..40960).
#define GEMM_LDS_BYTES 40960
template <bool OUT_F32>
__global__ __launch_bounds__(256) void gemm_bf16(const __bf16* __restrict__ A,
                                                 const __bf16* __restrict__ Bt,
                                                 void* __restrict__ Cout,
                                                 int M, int N, int K) {
  extern __shared__ char smem[];
  constexpr unsigned PITCH = 80, ABUF = 10240, BBASE = 20480;
  const int tid = threadIdx.x;
  const int lane = tid & 31;
  const int wave = tid >> 5;
  const int l16 = lane & 15, hi = lane >> 4;
  const int m0b = blockIdx.y * 128;
  const int n0b = blockIdx.x * 128;
  const int mw = m0b + (wave >> 2) * 64;   // wave's 64-row M range
  const int nw = n0b + (wave & 3) * 32;    // wave's 32-col N range

  // stage one 128x32 A slice and 128x32 B slice (64B rows) into buffer `buf`
  auto stage = [&](int kk, int buf) {
#pragma unroll
    for (int it = 0; it < 2; ++it) {
      const int idx = tid + it * 256;        // 0..511
      const int row = idx >> 2;
      const unsigned cb = (idx & 3) * 16u;
      async_b128(buf * ABUF + row * PITCH + cb,
                 (const char*)(A + (size_t)(m0b + row) * K + kk) + cb);
      async_b128(BBASE + buf * ABUF + row * PITCH + cb,
                 (const char*)(Bt + (size_t)(n0b + row) * K + kk) + cb);
    }
  };

  v8f acc[4][2];
#pragma unroll
  for (int i = 0; i < 4; ++i)
#pragma unroll
    for (int j = 0; j < 2; ++j) acc[i][j] = v8f{};

  stage(0, 0);
  int it = 0;
  for (int kk = 0; kk < K; kk += 32, ++it) {
    const int cur = it & 1;
    if (kk + 32 < K) {        // issue next slice's DMA, then counted-wait on current
      stage(kk + 32, cur ^ 1);
      wait_async4();
    } else {
      wait_async0();
    }
    __syncthreads();

    const char* ab = smem + cur * ABUF;
    const char* bb = smem + BBASE + cur * ABUF;
    v16bf af[4], bfr[2];
#pragma unroll
    for (int i = 0; i < 4; ++i) {
      const char* ar = ab + ((wave >> 2) * 64 + i * 16 + l16) * PITCH;
      af[i] = pack16(*(const v8bf*)(ar + 16 * hi), *(const v8bf*)(ar + 16 * hi + 32));
    }
#pragma unroll
    for (int j = 0; j < 2; ++j) {
      const char* br = bb + ((wave & 3) * 32 + j * 16 + l16) * PITCH;
      bfr[j] = pack16(*(const v8bf*)(br + 16 * hi), *(const v8bf*)(br + 16 * hi + 32));
    }
#pragma unroll
    for (int i = 0; i < 4; ++i)
#pragma unroll
      for (int j = 0; j < 2; ++j) acc[i][j] = wmma_bf16(af[i], bfr[j], acc[i][j]);
    __syncthreads();
  }

#pragma unroll
  for (int i = 0; i < 4; ++i)
#pragma unroll
    for (int j = 0; j < 2; ++j)
#pragma unroll
      for (int r = 0; r < 8; ++r) {
        int m = mw + i * 16 + r + 8 * hi;
        int n = nw + j * 16 + l16;
        if (OUT_F32)
          ((float*)Cout)[(size_t)m * N + n] = acc[i][j][r];
        else
          ((__bf16*)Cout)[(size_t)m * N + n] = (__bf16)acc[i][j][r];
      }
}

// ---------------- RoPE (in-place on bf16 q,k) ----------------
// gid = b(1) t(11) qi(1) h(4) i(6)  -> 2^23 threads
__global__ void rope_kernel(__bf16* __restrict__ qkv) {
  int gid = blockIdx.x * 256 + threadIdx.x;
  int i = gid & 63;
  int h = (gid >> 6) & 15;
  int qi = (gid >> 10) & 1;
  int t = (gid >> 11) & 2047;
  int b = gid >> 22;
  size_t base = (((size_t)(b * 2048 + t)) * 3 + qi) * 2048 + h * 128;
  // div[i] = exp(-ln(10000) * i / 64)
  float ang = (float)t * __expf((float)i * -0.14391156f);
  float sn, cs;
  __sincosf(ang, &sn, &cs);
  float t1 = (float)qkv[base + i];
  float t2 = (float)qkv[base + i + 64];
  qkv[base + i]      = (__bf16)(t1 * cs - t2 * sn);
  qkv[base + i + 64] = (__bf16)(t2 * cs + t1 * sn);
}

// ---------------- V transpose: vt[b][h][d][t] = v[b][t][h][d] ----------------
// gid = b(1) h(4) d(7) t(11)
__global__ void vtrans_kernel(const __bf16* __restrict__ qkv, __bf16* __restrict__ vt) {
  int gid = blockIdx.x * 256 + threadIdx.x;
  int t = gid & 2047;
  int d = (gid >> 11) & 127;
  int h = (gid >> 18) & 15;
  int b = gid >> 22;
  vt[(((size_t)(b * 16 + h)) * 128 + d) * 2048 + t] =
      qkv[(((size_t)(b * 2048 + t)) * 3 + 2) * 2048 + h * 128 + d];
}

// ---------------- Flash attention (causal, online softmax, WMMA) ----------------
// Grid: (T/64, B*H); block = 128 threads = 4 waves; wave owns 16 q rows.
// K and Vt chunks are double-buffered in LDS, moved by the Tensor Data Mover:
// wave 0 issues the K-tile descriptor, wave 1 the V-tile descriptor (1 instr each,
// EXEC-independent), with TDM hardware padding into the banked LDS pitches.
// Dynamic LDS layout (byte offsets, base 0):
//   K buffers : 2 x 32 rows x 272B  (0 .. 17408)     (256B data + 16B pad / row)
//   V buffers : 2 x 128 rows x 80B  (17408 .. 37888) ( 64B data + 16B pad / row)
//   P staging : 4 waves x 16 rows x 80B (37888 .. 43008)
#define FLASH_LDS_BYTES 43008
__global__ __launch_bounds__(128) void flash_attn(const __bf16* __restrict__ qkv,
                                                  const __bf16* __restrict__ vt,
                                                  __bf16* __restrict__ attnb) {
  extern __shared__ char smem[];
  constexpr int T = 2048, D = 128, R3 = 6144;
  constexpr unsigned KPITCH = 272, VPITCH = 80, PPITCH = 80;
  constexpr unsigned KBUF = 8704, VBASE = 17408, VBUF = 10240, PBASE = 37888;

  const int tid = threadIdx.x;
  const int lane = tid & 31;
  const int wave = tid >> 5;
  const int l16 = lane & 15, hi = lane >> 4;
  const int bh = blockIdx.y;  // b*16 + h
  const int b = bh >> 4, h = bh & 15;
  const int q0 = blockIdx.x * 64 + wave * 16;

  const __bf16* kbase = qkv + (size_t)b * T * R3 + 2048 + h * D;  // qkvi=1
  const __bf16* vtb = vt + (size_t)bh * D * T;

  // ---- TDM staging of one (K,V) chunk into LDS buffer `buf` ----
  auto stage = [&](int j, int buf) {
    if (wave == 0) {
      // K chunk: 32 rows x 128 elems, row stride R3; pad 4 dwords per 64 dwords
      tdm_load_2d(buf * KBUF, kbase + (size_t)(j * 32) * R3, 128, 32, R3, 5, 3);
    } else if (wave == 1) {
      // V chunk: 128 rows x 32 elems, row stride T; pad 4 dwords per 16 dwords
      tdm_load_2d(VBASE + buf * VBUF, vtb + j * 32, 32, 128, T, 3, 3);
    }
  };

  // Q fragments (A-layout), loaded once from global: q at qkvi=0
  const __bf16* qrow = qkv + ((size_t)(b * T + q0 + l16)) * R3 + h * D;
  v16bf qf[4];
#pragma unroll
  for (int kc = 0; kc < 4; ++kc) qf[kc] = load_a_frag(qrow, kc * 32, hi);

  v8f of[8];
#pragma unroll
  for (int f = 0; f < 8; ++f) of[f] = v8f{};
  float mi[8], li[8];
#pragma unroll
  for (int r = 0; r < 8; ++r) { mi[r] = -1e30f; li[r] = 0.0f; }

  const float scale = 0.08838834764831845f;  // 1/sqrt(128)
  const int jmax = blockIdx.x * 2 + 1;       // block-uniform trip count
  const int jlast = (q0 + 15) >> 5;          // this wave's causal bound
  __bf16* pb = (__bf16*)(smem + PBASE + wave * (16 * PPITCH));

  stage(0, 0);
  for (int j = 0; j <= jmax; ++j) {
    const int cur = j & 1;
    if (j < jmax) {           // issue next chunk's DMA first; TDM is in-order per wave
      stage(j + 1, cur ^ 1);
      __builtin_amdgcn_s_wait_tensorcnt(1);
    } else {
      __builtin_amdgcn_s_wait_tensorcnt(0);
    }
    __syncthreads();          // buf[cur] visible to all waves

    if (j <= jlast) {
      const char* kl = smem + cur * KBUF;
      const char* vl = smem + VBASE + cur * VBUF;

      // S = Q * K^T for 32 kv positions (two 16-col halves)
      v8f s[2];
#pragma unroll
      for (int nh = 0; nh < 2; ++nh) {
        v8f acc = v8f{};
        const char* krow = kl + (nh * 16 + l16) * KPITCH;
#pragma unroll
        for (int kc = 0; kc < 4; ++kc) {
          v16bf kf = pack16(*(const v8bf*)(krow + (kc * 32 + hi * 8) * 2), *(const v8bf*)(krow + (kc * 32 + hi * 8 + 16) * 2));
          acc = wmma_bf16(qf[kc], kf, acc);
        }
        s[nh] = acc;
      }

      // scale + causal mask; online softmax with DS_SWIZZLE xor reductions
      const int scol0 = j * 32 + l16;
#pragma unroll
      for (int r = 0; r < 8; ++r) {
        const int t = q0 + r + 8 * hi;
        float s0 = s[0][r] * scale; if (scol0 > t)      s0 = -1e30f;
        float s1 = s[1][r] * scale; if (scol0 + 16 > t) s1 = -1e30f;

        const float mx = red_max16(fmaxf(s0, s1));
        const float nm = fmaxf(mi[r], mx);
        const float al = __expf(mi[r] - nm);
        const float p0 = __expf(s0 - nm);
        const float p1 = __expf(s1 - nm);
        const float rs = red_sum16(p0 + p1);
        li[r] = li[r] * al + rs;
        mi[r] = nm;
#pragma unroll
        for (int f = 0; f < 8; ++f) of[f][r] *= al;
        // stage P (bf16) to per-wave LDS; same-wave DS ops are in-order
        pb[(r + 8 * hi) * (PPITCH / 2) + l16]      = (__bf16)p0;
        pb[(r + 8 * hi) * (PPITCH / 2) + 16 + l16] = (__bf16)p1;
      }

      // reload P in A-layout
      const char* prow = (const char*)pb + l16 * PPITCH;
      v16bf pf = pack16(*(const v8bf*)(prow + 32 * hi),
                        *(const v8bf*)(prow + 32 * hi + 16));

      // O += P(16x32) * V(32x128): Vt rows contiguous in kv -> direct B-frags
#pragma unroll
      for (int f = 0; f < 8; ++f) {
        v16bf vf = load_b16(vl + (f * 16 + l16) * VPITCH + hi * 32);
        of[f] = wmma_bf16(pf, vf, of[f]);
      }
    }
    __syncthreads();  // all reads of buf[cur] done before it is restaged
  }

  // finalize: O /= l; write [b][h][t][d] contiguous (matches reference's raw reshape)
#pragma unroll
  for (int r = 0; r < 8; ++r) {
    const float inv = 1.0f / li[r];
    const int t = q0 + r + 8 * hi;
    __bf16* orow = attnb + ((size_t)bh * T + t) * D;
#pragma unroll
    for (int f = 0; f < 8; ++f) orow[f * 16 + l16] = (__bf16)(of[f][r] * inv);
  }
}

// ---------------- host launch ----------------
extern "C" void kernel_launch(void* const* d_in, const int* in_sizes, int n_in,
                              void* d_out, int out_size, void* d_ws, size_t ws_size,
                              hipStream_t stream) {
  (void)in_sizes; (void)n_in; (void)out_size; (void)ws_size;
  const float* x    = (const float*)d_in[0];
  const float* Wqkv = (const float*)d_in[1];
  const float* Wout = (const float*)d_in[2];
  float* out = (float*)d_out;

  constexpr int B = 2, T = 2048, C = 2048;
  constexpr int M = B * T;        // 4096
  constexpr int K = C;            // 2048
  constexpr int N1 = 3 * C;       // 6144

  char* w = (char*)d_ws;
  __bf16* xb    = (__bf16*)(w);                          // 16 MB : x bf16 [M][K]
  __bf16* wqkvT = (__bf16*)(w + ((size_t)16 << 20));     // 24 MB : Wqkv^T [N1][K]
  __bf16* woutT = (__bf16*)(w + ((size_t)40 << 20));     //  8 MB : Wout^T [C][K]
  __bf16* qkv   = (__bf16*)(w + ((size_t)48 << 20));     // 48 MB : [B][T][3][H][D]
  __bf16* vt    = (__bf16*)(w + ((size_t)96 << 20));     // 16 MB : [B][H][D][T]
  __bf16* attnb = (__bf16*)(w + ((size_t)112 << 20));    // 16 MB : [B][H][T][D]

  cvt_bf16 <<<(M * K + 255) / 256, 256, 0, stream>>>(x, xb, M * K);
  cvtT_bf16<<<(K * N1 + 255) / 256, 256, 0, stream>>>(Wqkv, wqkvT, K, N1);
  cvtT_bf16<<<(K * C + 255) / 256, 256, 0, stream>>>(Wout, woutT, K, C);

  gemm_bf16<false><<<dim3(N1 / 128, M / 128), 256, GEMM_LDS_BYTES, stream>>>(
      xb, wqkvT, qkv, M, N1, K);

  rope_kernel  <<<(B * T * 2 * 16 * 64) / 256, 256, 0, stream>>>(qkv);
  vtrans_kernel<<<(B * 16 * 128 * T) / 256, 256, 0, stream>>>(qkv, vt);

  flash_attn<<<dim3(T / 64, B * 16), 128, FLASH_LDS_BYTES, stream>>>(qkv, vt, attnb);

  gemm_bf16<true><<<dim3(C / 128, M / 128), 256, GEMM_LDS_BYTES, stream>>>(
      attnb, woutT, out, M, C, K);
}
